// DOIT_39436389712066
// MI455X (gfx1250) — matrix operand order repaired and checked by reading, not compiled, not run on hardware
//
#include <hip/hip_runtime.h>
#include <hip/hip_bf16.h>

// ---------------------------------------------------------------------------
// MI455X (gfx1250) implementation.
// Compute-bound workload (~476 GFLOP vs ~310MB traffic -> ~1500 FLOP/byte),
// so all large GEMMs run on v_wmma_f32_16x16x32_f16 (f16 inputs, f32 accum).
// Weights are converted once to f16 and transposed to N-major so the WMMA
// B operand is two contiguous 16B loads per lane; A operand likewise.
// GEMM: wave tile 16Mx64N (4 independent WMMAs per K-step, A reused 4x),
// unconditional double-buffered loads in the main K loop (guards only in the
// single chunk-granular tail step), OOB rows clamped instead of zeroed so the
// hot loop is divergence-free. All memory ops go through explicit
// address_space(1) pointers so they lower to GLOBAL_* (LOADcnt/STOREcnt only)
// instead of FLAT_* (which would also tie up DScnt / the LDS path).
// ---------------------------------------------------------------------------

typedef _Float16 f16x8  __attribute__((ext_vector_type(8)));
typedef _Float16 f16x16 __attribute__((ext_vector_type(16)));
typedef float    f32x8  __attribute__((ext_vector_type(8)));

// global-address-space load/store helpers (force GLOBAL_* instructions)
template <typename T>
__device__ __forceinline__ T ldg(const T* p) {
  typedef __attribute__((address_space(1))) const T GT;
  return *(GT*)p;
}
template <typename T>
__device__ __forceinline__ void stg(T* p, T v) {
  typedef __attribute__((address_space(1))) T GT;
  *(GT*)p = v;
}

namespace cfg {
constexpr int B = 256, G = 20000, O = 3, P = 200, MAXG = 400, NC = 15000;
constexpr int DIM = 256, DEPTH = 4, HEADS = 8, DH = 64, MLP = 1024, OUT = 20;
constexpr int INNER = HEADS * DH;          // 512
constexpr int S     = 1 + P + 1;           // 202 tokens
constexpr long BS   = (long)B * S;         // 51712 rows
constexpr int KP    = MAXG * O;            // 1200
constexpr int KN    = NC * O;              // 45000
constexpr int QKVN  = 3 * INNER;           // 1536
constexpr int SP    = 208;                 // S padded to multiple of 16
}

// ----------------------------- wave reductions -----------------------------
__device__ __forceinline__ float wsum32(float v) {
#pragma unroll
  for (int m = 16; m > 0; m >>= 1) v += __shfl_xor(v, m, 32);
  return v;
}
__device__ __forceinline__ float wmax32(float v) {
#pragma unroll
  for (int m = 16; m > 0; m >>= 1) v = fmaxf(v, __shfl_xor(v, m, 32));
  return v;
}

__device__ __forceinline__ f16x16 mk16(const f16x8 lo, const f16x8 hi) {
  f16x16 r;
#pragma unroll
  for (int e = 0; e < 8; ++e) { r[e] = lo[e]; r[e + 8] = hi[e]; }
  return r;
}

// ------------------- generic batched strided WMMA GEMM ---------------------
// D[M,N] = act(alpha*A[M,K]@Bt[N,K]^T + bias) + res  -> f32 or f16 output.
// Wave tile 16Mx64N; block = 8 waves along M => 128M x 64N block tile.
// z = blockIdx.z split as z1 = z/nz2, z2 = z%nz2 for two batch strides.
// Requires K >= 32 and K % 8 == 0 (true for all call sites).
template <int ACT>  // 0 = none, 1 = exact GELU, 2 = ReLU
__global__ __launch_bounds__(256)
void gemm_wmma_f16(const _Float16* __restrict__ A, int lda, long sA1, long sA2,
                   const _Float16* __restrict__ Bt, int ldb, long sB1, long sB2,
                   const float* __restrict__ bias, long sBias,
                   const float* __restrict__ res,
                   float* __restrict__ outF32, _Float16* __restrict__ outF16,
                   int ldc, long sC1, long sC2,
                   int M, int N, int K, int nz2, float alpha) {
  const int z  = blockIdx.z;
  const int z1 = z / nz2, z2 = z - z1 * nz2;
  A  += (long)z1 * sA1 + (long)z2 * sA2;
  Bt += (long)z1 * sB1 + (long)z2 * sB2;
  const long co = (long)z1 * sC1 + (long)z2 * sC2;

  const int lane = threadIdx.x & 31;
  const int wv   = threadIdx.x >> 5;
  const int half = lane >> 4;
  const int l16  = lane & 15;

  const int m0 = (blockIdx.y * 8 + wv) * 16;
  if (m0 >= M) return;                      // wave-uniform: EXEC stays full
  const int n0 = blockIdx.x * 64;

  // A lane layout: elems 0..15 -> K = k0+16*half+e (contiguous in A row).
  int arow = m0 + l16; if (arow > M - 1) arow = M - 1;       // clamp (see hdr)
  const _Float16* pa = A + (long)arow * lda + 16 * half;

  // B lane layout: elems 0..15 -> K = k0+16*half+e (contiguous in Bt row).
  const _Float16* pb[4];
#pragma unroll
  for (int t = 0; t < 4; ++t) {
    int brow = n0 + 16 * t + l16;
    if (brow > N - 1) brow = N - 1;                          // clamp (see hdr)
    pb[t] = Bt + (long)brow * ldb + 16 * half;
  }

  f32x8 acc[4] = {{}, {}, {}, {}};
  const int K32 = K & ~31;

  // ---- software-pipelined main loop over K32 ----
  f16x8 A0 = ldg((const f16x8*)pa);
  f16x8 A1 = ldg((const f16x8*)(pa + 8));
  f16x8 B0[4], B1[4];
#pragma unroll
  for (int t = 0; t < 4; ++t) {
    B0[t] = ldg((const f16x8*)pb[t]);
    B1[t] = ldg((const f16x8*)(pb[t] + 8));
  }
#pragma unroll 2
  for (int k0 = 32; k0 < K32; k0 += 32) {
    pa += 32;
    const f16x8 nA0 = ldg((const f16x8*)pa);
    const f16x8 nA1 = ldg((const f16x8*)(pa + 8));
    f16x8 nB0[4], nB1[4];
#pragma unroll
    for (int t = 0; t < 4; ++t) {
      pb[t] += 32;
      nB0[t] = ldg((const f16x8*)pb[t]);
      nB1[t] = ldg((const f16x8*)(pb[t] + 8));
    }
    const f16x16 av = mk16(A0, A1);
#pragma unroll
    for (int t = 0; t < 4; ++t) {
      const f16x16 bv = mk16(B0[t], B1[t]);
      acc[t] = __builtin_amdgcn_wmma_f32_16x16x32_f16(false, av, false, bv,
                                                      (short)0, acc[t], false, false);
    }
    A0 = nA0; A1 = nA1;
#pragma unroll
    for (int t = 0; t < 4; ++t) { B0[t] = nB0[t]; B1[t] = nB1[t]; }
  }
  {
    const f16x16 av = mk16(A0, A1);
#pragma unroll
    for (int t = 0; t < 4; ++t) {
      const f16x16 bv = mk16(B0[t], B1[t]);
      acc[t] = __builtin_amdgcn_wmma_f32_16x16x32_f16(false, av, false, bv,
                                                      (short)0, acc[t], false, false);
    }
  }

  // ---- chunk-granular K tail (K % 32 != 0; K % 8 == 0) ----
  if (K32 < K) {
    const int ka0 = K32 + 16 * half;
    const int ka1 = ka0 + 8;
    f16x8 a0 = {}, a1 = {};
    if (ka0 < K) a0 = ldg((const f16x8*)(pa + 32));   // pa = base + (K32-32) + 16*half
    if (ka1 < K) a1 = ldg((const f16x8*)(pa + 40));
    const f16x16 av = mk16(a0, a1);
    const int kb0 = K32 + 16 * half;
    const int kb1 = kb0 + 8;
#pragma unroll
    for (int t = 0; t < 4; ++t) {
      f16x8 b0 = {}, b1 = {};
      if (kb0 < K) b0 = ldg((const f16x8*)(pb[t] + 32));  // base + (K32-32) + 16*half
      if (kb1 < K) b1 = ldg((const f16x8*)(pb[t] + 40));
      const f16x16 bv = mk16(b0, b1);
      acc[t] = __builtin_amdgcn_wmma_f32_16x16x32_f16(false, av, false, bv,
                                                      (short)0, acc[t], false, false);
    }
  }

  // ---- epilogue ----
#pragma unroll
  for (int t = 0; t < 4; ++t) {
    const int col = n0 + 16 * t + l16;
    if (col >= N) continue;
    const float bcol = bias ? ldg(bias + (long)z1 * sBias + col) : 0.0f;
#pragma unroll
    for (int r = 0; r < 8; ++r) {
      const int row = m0 + r + 8 * half;   // D layout: VGPR r -> M = r + 8*half
      if (row >= M) break;
      float v = acc[t][r] * alpha + bcol;
      if (ACT == 1) v = 0.5f * v * (1.0f + erff(v * 0.70710678118654752f));
      if (ACT == 2) v = fmaxf(v, 0.0f);
      if (res) v += ldg(res + (long)row * ldc + col);
      const long oi = co + (long)row * ldc + col;
      if (outF32) stg(outF32 + oi, v);
      if (outF16) stg(outF16 + oi, (_Float16)v);
    }
  }
}

// ------------------------------ LayerNorm ----------------------------------
// One wave per 256-wide row; 8 elems/lane; f16 output for the next GEMM.
__global__ __launch_bounds__(256)
void layernorm_f16(const float* __restrict__ in, long inStride,
                   const float* __restrict__ g, const float* __restrict__ b,
                   _Float16* __restrict__ out, long outStride, long nrows) {
  const int lane = threadIdx.x & 31, wv = threadIdx.x >> 5;
  const long row = (long)blockIdx.x * 8 + wv;
  if (row >= nrows) return;
  const f32x8 x = ldg((const f32x8*)(in + row * inStride) + lane);
  float s = 0.f;
#pragma unroll
  for (int e = 0; e < 8; ++e) s += x[e];
  const float mean = wsum32(s) * (1.0f / 256.0f);
  float q = 0.f;
#pragma unroll
  for (int e = 0; e < 8; ++e) { float d = x[e] - mean; q += d * d; }
  const float rstd = rsqrtf(wsum32(q) * (1.0f / 256.0f) + 1e-5f);
  const f32x8 gg = ldg((const f32x8*)g + lane);
  const f32x8 bb = ldg((const f32x8*)b + lane);
  f16x8 o;
#pragma unroll
  for (int e = 0; e < 8; ++e) o[e] = (_Float16)((x[e] - mean) * rstd * gg[e] + bb[e]);
  stg((f16x8*)(out + row * outStride) + lane, o);
}

// ------------------------------ softmax ------------------------------------
// One wave per attention row (len 202, stored padded to 208); in-place f16,
// writes zeros into the 6 pad columns so att@V can run with K=208 unguarded.
__global__ __launch_bounds__(256)
void softmax_rows(_Float16* __restrict__ att, long nrows) {
  using namespace cfg;
  const int lane = threadIdx.x & 31, wv = threadIdx.x >> 5;
  const long row = (long)blockIdx.x * 8 + wv;
  if (row >= nrows) return;
  _Float16* p = att + row * SP;
  float v[7];
  float mx = -3.0e38f;
#pragma unroll
  for (int t = 0; t < 7; ++t) {
    const int j = lane + 32 * t;
    v[t] = (j < S) ? (float)ldg(p + j) : -3.0e38f;
    mx = fmaxf(mx, v[t]);
  }
  mx = wmax32(mx);
  float s = 0.f;
#pragma unroll
  for (int t = 0; t < 7; ++t) {
    const int j = lane + 32 * t;
    if (j < S) { v[t] = expf(v[t] - mx); s += v[t]; }
  }
  s = wsum32(s);
  const float inv = 1.0f / s;
#pragma unroll
  for (int t = 0; t < 7; ++t) {
    const int j = lane + 32 * t;
    if (j < S)       stg(p + j, (_Float16)(v[t] * inv));
    else if (j < SP) stg(p + j, (_Float16)0.f);
  }
}

// ----------------------- weight convert + transpose ------------------------
// in: (batch, K, N) f32 row-major  ->  out: (batch, N, K) f16 row-major
__global__ __launch_bounds__(256)
void cvtT(const float* __restrict__ in, _Float16* __restrict__ outT, int K, int N) {
  const long total = (long)K * N;
  const long idx = (long)blockIdx.x * 256 + threadIdx.x;
  if (idx >= total) return;
  const long base = (long)blockIdx.z * total;
  const int k = (int)(idx / N);
  const int n = (int)(idx - (long)k * N);
  stg(outT + base + (long)n * K + k, (_Float16)ldg(in + base + idx));
}

// ------------------------------- gathers -----------------------------------
__global__ __launch_bounds__(256)
void gather_path(const float* __restrict__ x, const int* __restrict__ gidx,
                 _Float16* __restrict__ gxp) {
  using namespace cfg;
  const int p = blockIdx.z;
  const long f = (long)blockIdx.x * 256 + threadIdx.x;
  if (f >= (long)B * KP) return;
  const int b = (int)(f / KP);
  const int k = (int)(f - (long)b * KP);
  const int gi = k / O, o = k - gi * O;
  const int gene = ldg(gidx + p * MAXG + gi);
  stg(gxp + ((long)p * B + b) * KP + k, (_Float16)ldg(x + ((long)b * G + gene) * O + o));
}

__global__ __launch_bounds__(256)
void gather_nc(const float* __restrict__ x, const int* __restrict__ nidx,
               _Float16* __restrict__ gxn) {
  using namespace cfg;
  const long f = (long)blockIdx.x * 256 + threadIdx.x;
  if (f >= (long)B * KN) return;
  const int b = (int)(f / KN);
  const int j = (int)(f - (long)b * KN);
  const int jn = j / O, o = j - jn * O;
  stg(gxn + f, (_Float16)ldg(x + ((long)b * G + ldg(nidx + jn)) * O + o));
}

__global__ __launch_bounds__(256)
void set_cls(const float* __restrict__ cls, float* __restrict__ h) {
  using namespace cfg;
  const int i = blockIdx.x * 256 + threadIdx.x;
  if (i >= B * DIM) return;
  const int b = i >> 8, d = i & 255;
  stg(h + (long)b * S * DIM + d, ldg(cls + d));
}

// v^T materialization for att@V: vT[z=b*H+hd][d][j] = v[b][j][hd][d], pad j.
__global__ __launch_bounds__(256)
void transpose_v(const _Float16* __restrict__ qkv, _Float16* __restrict__ vT) {
  using namespace cfg;
  const int z = blockIdx.z;
  const int b = z >> 3, hd = z & 7;
  const int idx = blockIdx.x * 256 + threadIdx.x;
  if (idx >= DH * SP) return;
  const int d = idx / SP, j = idx - d * SP;
  _Float16 val = (_Float16)0.f;
  if (j < S) val = ldg(qkv + ((long)b * S + j) * QKVN + 2 * INNER + hd * DH + d);
  stg(vT + (long)z * DH * SP + idx, val);
}

// ------------------------------- launcher ----------------------------------
extern "C" void kernel_launch(void* const* d_in, const int* in_sizes, int n_in,
                              void* d_out, int out_size, void* d_ws, size_t ws_size,
                              hipStream_t stream) {
  using namespace cfg;
  (void)in_sizes; (void)n_in; (void)out_size; (void)ws_size;

  const float* x        = (const float*)d_in[0];
  const int*   gene_idx = (const int*)  d_in[1];
  const int*   nc_idx   = (const int*)  d_in[2];
  const float* W_path   = (const float*)d_in[3];
  const float* b_path   = (const float*)d_in[4];
  const float* W_nc     = (const float*)d_in[5];
  const float* b_nc     = (const float*)d_in[6];
  const float* cls_tok  = (const float*)d_in[7];
  const float* ln1_g    = (const float*)d_in[8];
  const float* ln1_b    = (const float*)d_in[9];
  const float* Wqkv     = (const float*)d_in[10];
  const float* Wo       = (const float*)d_in[11];
  const float* bo       = (const float*)d_in[12];
  const float* ln2_g    = (const float*)d_in[13];
  const float* ln2_b    = (const float*)d_in[14];
  const float* Wff1     = (const float*)d_in[15];
  const float* bff1     = (const float*)d_in[16];
  const float* Wff2     = (const float*)d_in[17];
  const float* bff2     = (const float*)d_in[18];
  const float* lnh_g    = (const float*)d_in[19];
  const float* lnh_b    = (const float*)d_in[20];
  const float* Wh1      = (const float*)d_in[21];
  const float* bh1      = (const float*)d_in[22];
  const float* Wh2      = (const float*)d_in[23];
  const float* bh2      = (const float*)d_in[24];
  float* out = (float*)d_out;

  // ----- scratch arena (~930 MB total) -----
  char* wptr = (char*)d_ws;
  auto alloc = [&](size_t bytes) -> char* {
    char* p = wptr; wptr += (bytes + 255) & ~(size_t)255; return p;
  };
  _Float16* WpathT = (_Float16*)alloc((size_t)P * DIM * KP * 2);
  _Float16* WncT   = (_Float16*)alloc((size_t)DIM * KN * 2);
  _Float16* WqkvT  = (_Float16*)alloc((size_t)DEPTH * QKVN * DIM * 2);
  _Float16* WoT    = (_Float16*)alloc((size_t)DEPTH * DIM * INNER * 2);
  _Float16* Wff1T  = (_Float16*)alloc((size_t)DEPTH * MLP * DIM * 2);
  _Float16* Wff2T  = (_Float16*)alloc((size_t)DEPTH * DIM * MLP * 2);
  _Float16* Wh1T   = (_Float16*)alloc((size_t)(4 * DIM) * DIM * 2);
  _Float16* Wh2T   = (_Float16*)alloc((size_t)OUT * (4 * DIM) * 2);
  _Float16* gxp    = (_Float16*)alloc((size_t)P * B * KP * 2);
  _Float16* gxn    = (_Float16*)alloc((size_t)B * KN * 2);
  float*    h      = (float*)   alloc((size_t)BS * DIM * 4);
  _Float16* y16    = (_Float16*)alloc((size_t)BS * DIM * 2);
  _Float16* qkv16  = (_Float16*)alloc((size_t)BS * QKVN * 2);
  _Float16* att16  = (_Float16*)alloc((size_t)B * HEADS * S * SP * 2);
  _Float16* vT16   = (_Float16*)alloc((size_t)B * HEADS * DH * SP * 2);
  _Float16* o16    = (_Float16*)alloc((size_t)BS * INNER * 2);
  _Float16* ff16   = (_Float16*)alloc((size_t)BS * MLP * 2);
  _Float16* z16    = (_Float16*)alloc((size_t)B * DIM * 2);
  _Float16* r16    = (_Float16*)alloc((size_t)B * 4 * DIM * 2);

  // ----- stage 0: weight f16 transpose-convert -----
  cvtT<<<dim3((KP * DIM + 255) / 256, 1, P),        256, 0, stream>>>(W_path, WpathT, KP, DIM);
  cvtT<<<dim3(((long)KN * DIM + 255) / 256, 1, 1),  256, 0, stream>>>(W_nc,   WncT,   KN, DIM);
  cvtT<<<dim3((DIM * QKVN + 255) / 256, 1, DEPTH),  256, 0, stream>>>(Wqkv,   WqkvT,  DIM, QKVN);
  cvtT<<<dim3((INNER * DIM + 255) / 256, 1, DEPTH), 256, 0, stream>>>(Wo,     WoT,    INNER, DIM);
  cvtT<<<dim3((DIM * MLP + 255) / 256, 1, DEPTH),   256, 0, stream>>>(Wff1,   Wff1T,  DIM, MLP);
  cvtT<<<dim3((MLP * DIM + 255) / 256, 1, DEPTH),   256, 0, stream>>>(Wff2,   Wff2T,  MLP, DIM);
  cvtT<<<dim3((DIM * 4 * DIM + 255) / 256, 1, 1),   256, 0, stream>>>(Wh1,    Wh1T,   DIM, 4 * DIM);
  cvtT<<<dim3((4 * DIM * OUT + 255) / 256, 1, 1),   256, 0, stream>>>(Wh2,    Wh2T,   4 * DIM, OUT);

  // ----- stage 1: gathers + cls token -----
  gather_path<<<dim3((B * KP + 255) / 256, 1, P), 256, 0, stream>>>(x, gene_idx, gxp);
  gather_nc<<<dim3(((long)B * KN + 255) / 256, 1, 1), 256, 0, stream>>>(x, nc_idx, gxn);
  set_cls<<<dim3((B * DIM + 255) / 256, 1, 1), 256, 0, stream>>>(cls_tok, h);

  // ----- stage 2: pathway + nc token GEMMs (write into token buffer h) -----
  gemm_wmma_f16<0><<<dim3(DIM / 64, B / 128, P), 256, 0, stream>>>(
      gxp, KP, (long)B * KP, 0,
      WpathT, KP, (long)DIM * KP, 0,
      b_path, DIM, nullptr,
      h + DIM, nullptr, S * DIM, DIM, 0,
      B, DIM, KP, 1, 1.0f);
  gemm_wmma_f16<0><<<dim3(DIM / 64, B / 128, 1), 256, 0, stream>>>(
      gxn, KN, 0, 0, WncT, KN, 0, 0, b_nc, 0, nullptr,
      h + (long)(S - 1) * DIM, nullptr, S * DIM, 0, 0,
      B, DIM, KN, 1, 1.0f);

  // ----- stage 3: transformer layers -----
  const unsigned mt  = (unsigned)((BS + 127) / 128);   // 404 M-blocks
  const unsigned mta = (unsigned)((S + 127) / 128);    // 2 M-blocks (attention)
  const float scale = 0.125f;                          // DH^-0.5
  for (int l = 0; l < DEPTH; ++l) {
    layernorm_f16<<<dim3((unsigned)((BS + 7) / 8)), 256, 0, stream>>>(
        h, DIM, ln1_g + l * DIM, ln1_b + l * DIM, y16, DIM, BS);
    gemm_wmma_f16<0><<<dim3(QKVN / 64, mt, 1), 256, 0, stream>>>(
        y16, DIM, 0, 0, WqkvT + (size_t)l * QKVN * DIM, DIM, 0, 0,
        nullptr, 0, nullptr,
        nullptr, qkv16, QKVN, 0, 0, (int)BS, QKVN, DIM, 1, 1.0f);
    transpose_v<<<dim3((DH * SP + 255) / 256, 1, B * HEADS), 256, 0, stream>>>(qkv16, vT16);
    // scores = scale * q @ k^T : Bt = k rows directly (already N-major)
    gemm_wmma_f16<0><<<dim3((S + 63) / 64, mta, B * HEADS), 256, 0, stream>>>(
        qkv16, QKVN, (long)S * QKVN, DH,
        qkv16 + INNER, QKVN, (long)S * QKVN, DH,
        nullptr, 0, nullptr,
        nullptr, att16, SP, (long)HEADS * S * SP, (long)S * SP,
        S, S, DH, HEADS, scale);
    softmax_rows<<<dim3((unsigned)(((long)B * HEADS * S + 7) / 8)), 256, 0, stream>>>(
        att16, (long)B * HEADS * S);
    // o = att @ v : K padded to 208 with zeros on both operands
    gemm_wmma_f16<0><<<dim3(1, mta, B * HEADS), 256, 0, stream>>>(
        att16, SP, (long)HEADS * S * SP, (long)S * SP,
        vT16, SP, (long)HEADS * DH * SP, (long)DH * SP,
        nullptr, 0, nullptr,
        nullptr, o16, INNER, (long)S * INNER, DH,
        S, DH, SP, HEADS, 1.0f);
    gemm_wmma_f16<0><<<dim3(DIM / 64, mt, 1), 256, 0, stream>>>(
        o16, INNER, 0, 0, WoT + (size_t)l * DIM * INNER, INNER, 0, 0,
        bo + l * DIM, 0, h,
        h, nullptr, DIM, 0, 0, (int)BS, DIM, INNER, 1, 1.0f);
    layernorm_f16<<<dim3((unsigned)((BS + 7) / 8)), 256, 0, stream>>>(
        h, DIM, ln2_g + l * DIM, ln2_b + l * DIM, y16, DIM, BS);
    gemm_wmma_f16<1><<<dim3(MLP / 64, mt, 1), 256, 0, stream>>>(   // exact GELU
        y16, DIM, 0, 0, Wff1T + (size_t)l * MLP * DIM, DIM, 0, 0,
        bff1 + l * MLP, 0, nullptr,
        nullptr, ff16, MLP, 0, 0, (int)BS, MLP, DIM, 1, 1.0f);
    gemm_wmma_f16<0><<<dim3(DIM / 64, mt, 1), 256, 0, stream>>>(
        ff16, MLP, 0, 0, Wff2T + (size_t)l * DIM * MLP, MLP, 0, 0,
        bff2 + l * DIM, 0, h,
        h, nullptr, DIM, 0, 0, (int)BS, DIM, MLP, 1, 1.0f);
  }

  // ----- stage 4: head (cls rows of h, stride S*DIM) -----
  layernorm_f16<<<dim3(B / 8), 256, 0, stream>>>(
      h, (long)S * DIM, lnh_g, lnh_b, z16, DIM, B);
  gemm_wmma_f16<2><<<dim3((4 * DIM) / 64, B / 128, 1), 256, 0, stream>>>(  // ReLU
      z16, DIM, 0, 0, Wh1T, DIM, 0, 0, bh1, 0, nullptr,
      nullptr, r16, 4 * DIM, 0, 0, B, 4 * DIM, DIM, 1, 1.0f);
  gemm_wmma_f16<0><<<dim3(1, B / 128, 1), 256, 0, stream>>>(
      r16, 4 * DIM, 0, 0, Wh2T, 4 * DIM, 0, 0, bh2, 0, nullptr,
      out, nullptr, OUT, 0, 0, B, OUT, 4 * DIM, 1, 1.0f);
}
